// PointerBlock_84567906058817
// MI455X (gfx1250) — matrix operand order repaired and checked by reading, not compiled, not run on hardware
//
#include <hip/hip_runtime.h>
#include <math.h>

// gfx1250 async global->LDS DMA path (ASYNCcnt), guarded so a toolchain without
// the builtins falls back to the VGPR-staged copy.
#if defined(__HIP_DEVICE_COMPILE__) && \
    __has_builtin(__builtin_amdgcn_global_load_async_to_lds_b128) && \
    __has_builtin(__builtin_amdgcn_s_wait_asynccnt)
#define USE_ASYNC_LDS 1
#else
#define USE_ASYNC_LDS 0
#endif

// ---------------- types ----------------
typedef __attribute__((ext_vector_type(8)))  float          v8f;
typedef __attribute__((ext_vector_type(8)))  __bf16         bf16x8;
typedef __attribute__((ext_vector_type(16))) __bf16         bf16x16;
typedef __attribute__((ext_vector_type(4)))  unsigned int   u32x4;
typedef __attribute__((ext_vector_type(4)))  int            i32x4;
typedef __attribute__((ext_vector_type(4)))  float          f32x4;
typedef __attribute__((ext_vector_type(4)))  unsigned short u16x4;

// ---------------- helpers ----------------
__device__ __forceinline__ unsigned short f2bf(float f) {
    unsigned int u = __float_as_uint(f);
    unsigned int r = u + 0x7FFFu + ((u >> 16) & 1u);   // round-to-nearest-even
    return (unsigned short)(r >> 16);
}
__device__ __forceinline__ float bf2f(unsigned short s) {
    return __uint_as_float((unsigned int)s << 16);
}
__device__ __forceinline__ float gelu_erf(float x) {
    return 0.5f * x * (1.0f + erff(x * 0.70710678118654752440f));
}

#if USE_ASYNC_LDS
__device__ __forceinline__ void async_cp16(unsigned short* lds, const unsigned short* g) {
    // signature (per hipcc diagnostic): params are int4* (global src, LDS dst), imm offset, imm cpol
    __builtin_amdgcn_global_load_async_to_lds_b128((i32x4*)g, (i32x4*)lds, 0, 0);
}
#endif

// ---------------- elementwise f32 -> bf16 (vectorized b128->b64) ----------------
__global__ __launch_bounds__(256)
void k_cvt_bf16_v4(const float* __restrict__ in, unsigned short* __restrict__ out, int n4) {
    for (int i = blockIdx.x * blockDim.x + threadIdx.x; i < n4; i += gridDim.x * blockDim.x) {
        f32x4 v = ((const f32x4*)in)[i];
        u16x4 o;
        o.x = f2bf(v.x); o.y = f2bf(v.y); o.z = f2bf(v.z); o.w = f2bf(v.w);
        ((u16x4*)out)[i] = o;
    }
}

// ---------------- transpose + convert: W[R][C] f32 -> Wt[C][R] bf16 ----------------
__global__ __launch_bounds__(256)
void k_transpose_bf16(const float* __restrict__ in, unsigned short* __restrict__ out,
                      int R, int C) {
    __shared__ unsigned short tile[32][33];
    const int tx = threadIdx.x;           // 0..31
    const int ty = threadIdx.y;           // 0..7
    const int r0 = blockIdx.y * 32;
    const int c0 = blockIdx.x * 32;
#pragma unroll
    for (int i = 0; i < 32; i += 8)
        tile[ty + i][tx] = f2bf(in[(size_t)(r0 + ty + i) * C + (c0 + tx)]);
    __syncthreads();
#pragma unroll
    for (int i = 0; i < 32; i += 8)
        out[(size_t)(c0 + ty + i) * R + (r0 + tx)] = tile[tx][ty + i];
}

// ---------------- logits / sigmoid / pointer targets ----------------
__global__ __launch_bounds__(256)
void k_logits(const unsigned short* __restrict__ hidden,   // [M][512] bf16
              const float* __restrict__ W_re2, const float* __restrict__ b_re2,
              float* __restrict__ strength_out, float* __restrict__ target_out_f,
              int* __restrict__ target_out_i, int M, int seqN) {
    const int tok  = blockIdx.x * 8 + (threadIdx.x >> 5);  // one wave32 per token
    const int lane = threadIdx.x & 31;
    if (tok >= M) return;
    const unsigned short* hrow = hidden + (size_t)tok * 512;
    float s = 0.f;
#pragma unroll
    for (int j = lane; j < 512; j += 32)
        s += bf2f(hrow[j]) * W_re2[j];
#pragma unroll
    for (int off = 16; off > 0; off >>= 1)
        s += __shfl_xor(s, off, 32);
    if (lane == 0) {
        float logit = s + b_re2[0];
        float p = 1.0f / (1.0f + expf(-logit));
        float t = rintf(p * (float)(seqN - 1));            // half-to-even like jnp.round
        int ti = (int)t;
        ti = ti < 0 ? 0 : (ti > seqN - 1 ? seqN - 1 : ti);
        strength_out[tok] = p;
        target_out_f[tok] = (float)ti;
        target_out_i[tok] = ti;
    }
}

// ---------------- fold bias: bc[n] = sum_b b_rt2[b] * W_o[b][n] ----------------
__global__ __launch_bounds__(256)
void k_bias_fold(const float* __restrict__ b_rt2, const float* __restrict__ W_o,
                 float* __restrict__ bc, int d) {
    const int n = blockIdx.x * blockDim.x + threadIdx.x;
    if (n >= d) return;
    float s = 0.f;
    for (int b = 0; b < d; ++b)
        s += b_rt2[b] * W_o[(size_t)b * d + n];
    bc[n] = s;
}

// ---------------- WMMA GEMM: C[m][n] = sum_k A[m][k] * Bt[n][k] ----------------
// Block tile 128x128, 8 wave32 waves (2x4), wave tile 64x32 (4x2 of 16x16),
// BK=32 with v_wmma_f32_16x16x32_bf16. Double-buffered LDS fed by async DMA.
// Optional row-gather on A for k >= kHalf (pointer half of relation_input).
enum { EPI_BF16 = 0, EPI_GELU_BF16 = 1, EPI_BIAS_F32 = 2 };

template <int EPI, bool GATHER>
__global__ __launch_bounds__(256)
void k_gemm(const unsigned short* __restrict__ A, int lda,
            const unsigned short* __restrict__ Bt, int ldb,
            void* __restrict__ Out, int ldo,
            const float* __restrict__ bias,
            int K, int kHalf,
            const int* __restrict__ tgt, int seqN) {
    constexpr int BM = 128, BN = 128, BK = 32, LDT = 48;   // LDT: 16B-aligned pad
    __shared__ __attribute__((aligned(16))) unsigned short sA[2][BM * LDT];
    __shared__ __attribute__((aligned(16))) unsigned short sB[2][BN * LDT];

    const int tid  = threadIdx.x;
    const int lane = tid & 31;
    const int wave = tid >> 5;
    const int wm   = wave >> 2;   // 0..1 : 64-row slab
    const int wn   = wave & 3;    // 0..3 : 32-col slab
    const int lr   = lane & 15;
    const int kh   = lane >> 4;

    const int blockM = blockIdx.y * BM;
    const int blockN = blockIdx.x * BN;

    // loader: each thread owns two 16B chunks of A and two of B per K-step
    const int arow = tid >> 2;          // 0..63
    const int acol = (tid & 3) * 8;     // 0,8,16,24
    const int gsrc0 = blockM + arow;
    const int gsrc1 = gsrc0 + 64;
    int grow0 = gsrc0, grow1 = gsrc1;
    if (GATHER) {
        grow0 = (gsrc0 / seqN) * seqN + tgt[gsrc0];
        grow1 = (gsrc1 / seqN) * seqN + tgt[gsrc1];
    }
    const int brow0 = blockN + arow;
    const int brow1 = brow0 + 64;

    v8f acc[4][2] = {};

    auto srcA0 = [&](int k0) -> const unsigned short* {
        if (GATHER && k0 >= kHalf) return A + (size_t)grow0 * lda + (k0 - kHalf) + acol;
        return A + (size_t)gsrc0 * lda + k0 + acol;
    };
    auto srcA1 = [&](int k0) -> const unsigned short* {
        if (GATHER && k0 >= kHalf) return A + (size_t)grow1 * lda + (k0 - kHalf) + acol;
        return A + (size_t)gsrc1 * lda + k0 + acol;
    };

    auto compute_tile = [&](const unsigned short* cA, const unsigned short* cB) {
        bf16x16 af[4], bm[2];
        // A frag: lane group kh holds k in [kh*8, kh*8+8) and [16+kh*8, ...)
#pragma unroll
        for (int i = 0; i < 4; ++i) {
            const unsigned short* p = cA + (wm * 64 + i * 16 + lr) * LDT + kh * 8;
            bf16x8 lo = *(const bf16x8*)p;
            bf16x8 hi = *(const bf16x8*)(p + 16);
            af[i] = __builtin_shufflevector(lo, hi, 0,1,2,3,4,5,6,7,8,9,10,11,12,13,14,15);
        }
        // B frag: lane group kh holds k in [kh*8, kh*8+8) and [16+kh*8, ...)
#pragma unroll
        for (int j = 0; j < 2; ++j) {
            const unsigned short* p = cB + (wn * 32 + j * 16 + lr) * LDT + kh * 8;
            bf16x8 lo = *(const bf16x8*)p;
            bf16x8 hi = *(const bf16x8*)(p + 16);
            bm[j] = __builtin_shufflevector(lo, hi, 0,1,2,3,4,5,6,7,8,9,10,11,12,13,14,15);
        }
#pragma unroll
        for (int i = 0; i < 4; ++i)
#pragma unroll
            for (int j = 0; j < 2; ++j)
                acc[i][j] = __builtin_amdgcn_wmma_f32_16x16x32_bf16(
                    false, af[i], false, bm[j], (short)0, acc[i][j], false, false);
    };

#if USE_ASYNC_LDS
    // software-pipelined: issue async DMA for next tile, wait only current 4 copies
    auto stage = [&](int k0, int b) {
        unsigned short* dA = &sA[b][arow * LDT + acol];
        unsigned short* dB = &sB[b][arow * LDT + acol];
        async_cp16(dA,            srcA0(k0));
        async_cp16(dA + 64 * LDT, srcA1(k0));
        async_cp16(dB,            Bt + (size_t)brow0 * ldb + k0 + acol);
        async_cp16(dB + 64 * LDT, Bt + (size_t)brow1 * ldb + k0 + acol);
    };
    stage(0, 0);
    int buf = 0;
    for (int k0 = 0; k0 + BK < K; k0 += BK, buf ^= 1) {
        stage(k0 + BK, buf ^ 1);
        __builtin_amdgcn_s_wait_asynccnt(4);   // current tile's 4 copies done (in-order)
        __syncthreads();
        compute_tile(&sA[buf][0], &sB[buf][0]);
        __syncthreads();                       // readers done before buf is re-targeted
    }
    __builtin_amdgcn_s_wait_asynccnt(0);
    __syncthreads();
    compute_tile(&sA[buf][0], &sB[buf][0]);
#else
    for (int k0 = 0; k0 < K; k0 += BK) {
        __syncthreads();
        *(u32x4*)(&sA[0][arow * LDT + acol])        = *(const u32x4*)srcA0(k0);
        *(u32x4*)(&sA[0][(arow + 64) * LDT + acol]) = *(const u32x4*)srcA1(k0);
        *(u32x4*)(&sB[0][arow * LDT + acol])        = *(const u32x4*)(Bt + (size_t)brow0 * ldb + k0 + acol);
        *(u32x4*)(&sB[0][(arow + 64) * LDT + acol]) = *(const u32x4*)(Bt + (size_t)brow1 * ldb + k0 + acol);
        __syncthreads();
        compute_tile(&sA[0][0], &sB[0][0]);
    }
#endif

    // epilogue: lane lr = N column within tile, rows m = kh*8 + r
#pragma unroll
    for (int i = 0; i < 4; ++i) {
#pragma unroll
        for (int j = 0; j < 2; ++j) {
            const int n = blockN + wn * 32 + j * 16 + lr;
            const float bv = (EPI == EPI_BF16) ? 0.0f : bias[n];
#pragma unroll
            for (int r = 0; r < 8; ++r) {
                const int m = blockM + wm * 64 + i * 16 + kh * 8 + r;
                float x = acc[i][j][r] + bv;
                if (EPI == EPI_GELU_BF16) x = gelu_erf(x);
                if (EPI == EPI_BIAS_F32)
                    ((float*)Out)[(size_t)m * ldo + n] = x;
                else
                    ((unsigned short*)Out)[(size_t)m * ldo + n] = f2bf(x);
            }
        }
    }
}

// ---------------- driver ----------------
extern "C" void kernel_launch(void* const* d_in, const int* in_sizes, int n_in,
                              void* d_out, int out_size, void* d_ws, size_t ws_size,
                              hipStream_t stream) {
    const int Bb = 4, Nn = 4096, d = 1024, Kre = 512;
    const int M = Bb * Nn;                 // 16384 tokens

    const float* h     = (const float*)d_in[0];
    const float* W_re1 = (const float*)d_in[1];
    const float* b_re1 = (const float*)d_in[2];
    const float* W_re2 = (const float*)d_in[3];
    const float* b_re2 = (const float*)d_in[4];
    const float* W_v   = (const float*)d_in[5];
    const float* W_rt1 = (const float*)d_in[6];
    const float* b_rt1 = (const float*)d_in[7];
    const float* W_rt2 = (const float*)d_in[8];
    const float* b_rt2 = (const float*)d_in[9];
    const float* W_o   = (const float*)d_in[10];

    float* z_out   = (float*)d_out;                     // [M][d]
    float* tgt_out = z_out + (size_t)M * d;             // [M] (int values as f32)
    float* str_out = tgt_out + M;                       // [M]

    // workspace carve (256B aligned)
    char* ws = (char*)d_ws; size_t off = 0;
    auto carve = [&](size_t bytes) -> char* {
        char* p = ws + off; off = (off + bytes + 255) & ~(size_t)255; return p;
    };
    unsigned short* hbf    = (unsigned short*)carve((size_t)M * d * 2);   // reused as R
    unsigned short* hidden = (unsigned short*)carve((size_t)M * Kre * 2);
    unsigned short* Vbf    = (unsigned short*)carve((size_t)M * d * 2);
    unsigned short* Wre1t  = (unsigned short*)carve((size_t)Kre * d * 2); // [512][1024]
    unsigned short* Wvt    = (unsigned short*)carve((size_t)d * d * 2);   // [1024][1024]
    unsigned short* Wrt1t  = (unsigned short*)carve((size_t)d * 2 * d * 2);// [1024][2048]
    unsigned short* Wrt2b  = (unsigned short*)carve((size_t)d * d * 2);   // native layout
    unsigned short* Wot    = (unsigned short*)carve((size_t)d * d * 2);   // W_o^T
    unsigned short* Wct    = (unsigned short*)carve((size_t)d * d * 2);   // (W_rt2@W_o)^T
    float*          bcv    = (float*)carve((size_t)d * 4);
    int*            tgt_i  = (int*)carve((size_t)M * 4);
    unsigned short* Rbf    = hbf;  // reuse: h_bf16 dead after V GEMM

    dim3 blk256(256);

    // 0) precision conversions / weight transposes
    k_cvt_bf16_v4<<<4096, blk256, 0, stream>>>(h, hbf, M * d / 4);
    k_cvt_bf16_v4<<<1024, blk256, 0, stream>>>(W_rt2, Wrt2b, d * d / 4);
    dim3 tb(32, 8);
    k_transpose_bf16<<<dim3(Kre / 32, d / 32),   tb, 0, stream>>>(W_re1, Wre1t, d, Kre);
    k_transpose_bf16<<<dim3(d / 32, d / 32),     tb, 0, stream>>>(W_v,   Wvt,   d, d);
    k_transpose_bf16<<<dim3(d / 32, 2 * d / 32), tb, 0, stream>>>(W_rt1, Wrt1t, 2 * d, d);
    k_transpose_bf16<<<dim3(d / 32, d / 32),     tb, 0, stream>>>(W_o,   Wot,   d, d);

    // 1) hidden = gelu(h @ W_re1 + b_re1)            M x 512, K=1024
    k_gemm<EPI_GELU_BF16, false><<<dim3(Kre / 128, M / 128), blk256, 0, stream>>>(
        hbf, d, Wre1t, d, hidden, Kre, b_re1, d, d, nullptr, Nn);

    // 2) logits -> strength / pointer targets
    k_logits<<<M / 8, blk256, 0, stream>>>(hidden, W_re2, b_re2, str_out, tgt_out, tgt_i, M, Nn);

    // 3) V = h @ W_v                                  M x 1024, K=1024
    k_gemm<EPI_BF16, false><<<dim3(d / 128, M / 128), blk256, 0, stream>>>(
        hbf, d, Wvt, d, Vbf, d, nullptr, d, d, nullptr, Nn);

    // 4) Wct = (W_rt2 @ W_o)^T  via  W_o^T @ W_rt2^T  (folds last two GEMMs into one)
    k_gemm<EPI_BF16, false><<<dim3(d / 128, d / 128), blk256, 0, stream>>>(
        Wot, d, Wrt2b, d, Wct, d, nullptr, d, d, nullptr, Nn);
    k_bias_fold<<<d / 256, blk256, 0, stream>>>(b_rt2, W_o, bcv, d);

    // 5) R = gelu([V | gather(V)] @ W_rt1 + b_rt1)    M x 1024, K=2048 (gather half)
    k_gemm<EPI_GELU_BF16, true><<<dim3(d / 128, M / 128), blk256, 0, stream>>>(
        Vbf, d, Wrt1t, 2 * d, Rbf, d, b_rt1, 2 * d, d, tgt_i, Nn);

    // 6) z = R @ Wc + bc                              M x 1024, K=1024, f32 out
    k_gemm<EPI_BIAS_F32, false><<<dim3(d / 128, M / 128), blk256, 0, stream>>>(
        Rbf, d, Wct, d, z_out, d, bcv, d, d, nullptr, Nn);

    (void)in_sizes; (void)n_in; (void)out_size; (void)ws_size;
}
